// ROIViTExtractor_85143431676313
// MI455X (gfx1250) — matrix operand order repaired and checked by reading, not compiled, not run on hardware
//
#include <hip/hip_runtime.h>
#include <math.h>

typedef __attribute__((ext_vector_type(16))) _Float16 v16h;
typedef __attribute__((ext_vector_type(8)))  _Float16 v8h;
typedef __attribute__((ext_vector_type(8)))  float    v8f;

#define L_LAYERS 12
#define HEADS    12
#define DIM      768
#define HEAD_DIM 64
#define MLP_DIM  3072
#define SEQ      50
#define N_ROI    64
#define N_PATCH  49
#define TOKENS   (SEQ * N_ROI)   // 3200
#define LN_EPS   1e-5f

// ---------------- WMMA GEMM:  C[M,N] = epilogue(A[M,K] * W[N,K]^T + bias) ----------
// Block tile 128x64, BK=32; 256 threads = 8 waves in a 4(M) x 2(N) grid;
// each wave computes a 32x32 tile = 2x2 WMMA fragments (A/B fragment reuse 2x).
// Staging: thread -> contiguous 16 floats of one A row (4x b128 loads, 2x b128 LDS
// stores) / 8 floats of one B row. act: 0=none 1=relu 2=quickgelu; residual: C += .
#define BM 128
#define BN 64
#define BK 32
#define TS 40   // LDS row stride in halves (80B: 16B-aligned rows, conflict-free)

// A-fragment (16x32 f16, ISA 7.12.2): lanes 0-15 row M=L, K in {0..7}U{16..23};
// lanes 16-31 row M=L-16, K in {8..15}U{24..31}  -> two contiguous 16B chunks/lane
__device__ inline v16h frag_a_lds(const _Float16* base, int lane) {
  const int row = lane & 15;
  const int hi  = lane >> 4;
  const _Float16* p = base + row * TS;
  v16h f;
#pragma unroll
  for (int e = 0; e < 16; ++e) {
    int k = (e < 8) ? ((hi ? 8 : 0) + e) : ((hi ? 24 : 16) + (e - 8));
    f[e] = p[k];
  }
  return f;
}
// B-fragment (32x16 f16): lane L holds column n=L&15; lanes 0-15: K in {0..7}U{16..23},
// lanes 16-31: K in {8..15}U{24..31} (same K order as the A-fragment).
__device__ inline v16h frag_b_lds(const _Float16* base, int lane) {
  const _Float16* p = base + (lane & 15) * TS + ((lane >> 4) ? 8 : 0);
  v16h f;
#pragma unroll
  for (int e = 0; e < 16; ++e) f[e] = p[(e < 8) ? e : (e + 8)];
  return f;
}

__global__ void __launch_bounds__(256)
gemm_wmma(const float* __restrict__ A, const float* __restrict__ W,
          const float* __restrict__ bias, float* __restrict__ C,
          int M, int N, int K, int act, int residual)
{
  __shared__ alignas(16) _Float16 As[BM * TS];   // 10240 B
  __shared__ alignas(16) _Float16 Bs[BN * TS];   //  5120 B
  const int tid  = threadIdx.x;
  const int lane = tid & 31;
  const int wave = tid >> 5;     // 8 waves
  const int wm   = wave >> 1;    // 0..3  (32-row wave tile)
  const int wn   = wave & 1;     // 0..1  (32-col wave tile)
  const int m0   = blockIdx.y * BM;
  const int n0   = blockIdx.x * BN;

  // static staging assignment: exact cover, no loops, no divergence
  const int ar = tid >> 1;           // A row 0..127
  const int ac = (tid & 1) << 4;     // A col offset 0 or 16
  const int br = tid >> 2;           // B row 0..63
  const int bc = (tid & 3) << 3;     // B col offset 0,8,16,24
  const int agr = m0 + ar;
  const long abase = (long)agr * K + ac;
  const long bbase = (long)(n0 + br) * K + bc;

  v8f acc00 = {}, acc01 = {}, acc10 = {}, acc11 = {};

  for (int k0 = 0; k0 < K; k0 += BK) {
    // ---- stage A: 16 contiguous floats -> 16 halves ----
    float fa[16];
#pragma unroll
    for (int j = 0; j < 16; ++j) fa[j] = 0.f;
    if (agr < M) {
      const float4* ap = (const float4*)(A + abase + k0);
      *(float4*)(fa + 0)  = ap[0];
      *(float4*)(fa + 4)  = ap[1];
      *(float4*)(fa + 8)  = ap[2];
      *(float4*)(fa + 12) = ap[3];
      if (k0 + BK < K) __builtin_prefetch(ap + 8, 0, 1);  // next K tile
    }
    v8h ha0, ha1;
#pragma unroll
    for (int j = 0; j < 8; ++j) { ha0[j] = (_Float16)fa[j]; ha1[j] = (_Float16)fa[j + 8]; }
    *(v8h*)(As + ar * TS + ac)     = ha0;
    *(v8h*)(As + ar * TS + ac + 8) = ha1;

    // ---- stage B: 8 contiguous floats -> 8 halves ----
    float fb[8];
    {
      const float4* bp = (const float4*)(W + bbase + k0);
      *(float4*)(fb + 0) = bp[0];
      *(float4*)(fb + 4) = bp[1];
      if (k0 + BK < K) __builtin_prefetch(bp + 8, 0, 1);
    }
    v8h hb;
#pragma unroll
    for (int j = 0; j < 8; ++j) hb[j] = (_Float16)fb[j];
    *(v8h*)(Bs + br * TS + bc) = hb;

    __syncthreads();
    v16h a0 = frag_a_lds(As + (wm * 32 + 0)  * TS, lane);
    v16h a1 = frag_a_lds(As + (wm * 32 + 16) * TS, lane);
    v16h b0 = frag_b_lds(Bs + (wn * 32 + 0)  * TS, lane);
    v16h b1 = frag_b_lds(Bs + (wn * 32 + 16) * TS, lane);
    acc00 = __builtin_amdgcn_wmma_f32_16x16x32_f16(false, a0, false, b0, (short)0, acc00, false, false);
    acc01 = __builtin_amdgcn_wmma_f32_16x16x32_f16(false, a0, false, b1, (short)0, acc01, false, false);
    acc10 = __builtin_amdgcn_wmma_f32_16x16x32_f16(false, a1, false, b0, (short)0, acc10, false, false);
    acc11 = __builtin_amdgcn_wmma_f32_16x16x32_f16(false, a1, false, b1, (short)0, acc11, false, false);
    __syncthreads();
  }

  // epilogue: C/D layout = lane&15 -> column, VGPR r -> row (lanes>=16: row+8)
  v8f* accs[4] = { &acc00, &acc01, &acc10, &acc11 };
#pragma unroll
  for (int ti = 0; ti < 2; ++ti) {       // sub-tile row
#pragma unroll
    for (int tj = 0; tj < 2; ++tj) {     // sub-tile col
      v8f& acc = *accs[ti * 2 + tj];
      const int gcol = n0 + wn * 32 + tj * 16 + (lane & 15);
      const float bv = bias ? bias[gcol] : 0.f;
#pragma unroll
      for (int r = 0; r < 8; ++r) {
        int mloc = (lane < 16) ? r : (r + 8);
        int grow = m0 + wm * 32 + ti * 16 + mloc;
        if (grow < M) {
          float val = acc[r] + bv;
          if (act == 1)      val = fmaxf(val, 0.f);
          else if (act == 2) // QuickGELU: x * sigmoid(1.702 x), v_rcp fast path
            val = val * __builtin_amdgcn_rcpf(1.f + __expf(-1.702f * val));
          long o = (long)grow * N + gcol;
          if (residual) val += C[o];
          C[o] = val;
        }
      }
    }
  }
}

// ---------------- LayerNorm over last dim (DIM), one block per row ----------------
__global__ void __launch_bounds__(256)
layernorm_k(const float* __restrict__ x, float* __restrict__ y,
            const float* __restrict__ g, const float* __restrict__ b)
{
  __shared__ float red[256];
  const int row = blockIdx.x;
  const float* xr = x + (long)row * DIM;
  float s = 0.f;
  for (int d = threadIdx.x; d < DIM; d += 256) s += xr[d];
  red[threadIdx.x] = s; __syncthreads();
  for (int o = 128; o > 0; o >>= 1) {
    if (threadIdx.x < o) red[threadIdx.x] += red[threadIdx.x + o];
    __syncthreads();
  }
  const float mean = red[0] * (1.f / DIM);
  __syncthreads();
  float s2 = 0.f;
  for (int d = threadIdx.x; d < DIM; d += 256) { float t = xr[d] - mean; s2 += t * t; }
  red[threadIdx.x] = s2; __syncthreads();
  for (int o = 128; o > 0; o >>= 1) {
    if (threadIdx.x < o) red[threadIdx.x] += red[threadIdx.x + o];
    __syncthreads();
  }
  const float rstd = rsqrtf(red[0] * (1.f / DIM) + LN_EPS);
  for (int d = threadIdx.x; d < DIM; d += 256)
    y[(long)row * DIM + d] = (xr[d] - mean) * rstd * g[d] + b[d];
}

// -------- diagonal-softmax attention: o[l] = softmax(scores_l)[l] * v[l] ----------
// block = (roi n, head h), 128 threads = 4 waves, wave per seq-row stride 4
__global__ void __launch_bounds__(128)
attn_diag_k(const float* __restrict__ q, const float* __restrict__ k,
            const float* __restrict__ v, const float* __restrict__ Mov,
            float* __restrict__ o)
{
  __shared__ float qs[SEQ * 65];
  __shared__ float ks[SEQ * 65];
  __shared__ float mrow[N_PATCH];
  const int n = blockIdx.x, h = blockIdx.y;
  const int tid = threadIdx.x, lane = tid & 31, wave = tid >> 5;
  const float scale = 0.125f;  // 1/sqrt(64)

  for (int idx = tid; idx < SEQ * HEAD_DIM; idx += 128) {
    int s = idx >> 6, d = idx & 63;
    long g = ((long)s * N_ROI + n) * DIM + h * HEAD_DIM + d;
    qs[s * 65 + d] = q[g];
    ks[s * 65 + d] = k[g];
  }
  for (int idx = tid; idx < N_PATCH; idx += 128) mrow[idx] = Mov[n * N_PATCH + idx];
  __syncthreads();

  for (int l = wave; l < SEQ; l += 4) {
    const float* ql = qs + l * 65;
    float sc[2]; float mx = -1e30f;
#pragma unroll
    for (int c = 0; c < 2; ++c) {
      int t = lane + 32 * c;
      float sv = -1e30f;
      if (t < SEQ) {
        float dot = 0.f;
        const float* kt = ks + t * 65;
        for (int d = 0; d < HEAD_DIM; ++d) dot += ql[d] * kt[d];
        float bias = 0.f;
        if (l == 0 && t >= 1)      bias = mrow[t - 1];
        else if (t == 0 && l >= 1) bias = mrow[l - 1];
        sv = dot * scale + bias;
      }
      sc[c] = sv;
      mx = fmaxf(mx, sv);
    }
    for (int off = 16; off > 0; off >>= 1) mx = fmaxf(mx, __shfl_xor(mx, off));
    float sum = 0.f;
#pragma unroll
    for (int c = 0; c < 2; ++c) if (lane + 32 * c < SEQ) sum += __expf(sc[c] - mx);
    for (int off = 16; off > 0; off >>= 1) sum += __shfl_xor(sum, off);
    // diagonal score (diag bias is 0)
    float ddot = 0.f;
    const float* kl = ks + l * 65;
    for (int d = 0; d < HEAD_DIM; ++d) ddot += ql[d] * kl[d];
    const float a = __expf(ddot * scale - mx) * __builtin_amdgcn_rcpf(sum);
    for (int d = lane; d < HEAD_DIM; d += 32) {
      long g = ((long)l * N_ROI + n) * DIM + h * HEAD_DIM + d;
      o[g] = a * v[g];
    }
  }
}

// ---------------- small setup kernels ----------------
__global__ void im2col_k(const float* __restrict__ frame, float* __restrict__ col) {
  int idx = blockIdx.x * 256 + threadIdx.x;
  if (idx >= N_PATCH * 3072) return;
  int p = idx / 3072, kk = idx % 3072;
  int c = kk >> 10, r = (kk >> 5) & 31, cc = kk & 31;
  int pr = p / 7, pc = p % 7;
  col[idx] = frame[c * 224 * 224 + (pr * 32 + r) * 224 + (pc * 32 + cc)];
}

__global__ void overlap_k(const float* __restrict__ bb, float* __restrict__ Mov) {
  int idx = blockIdx.x * 256 + threadIdx.x;
  if (idx >= N_ROI * N_PATCH) return;
  int n = idx / N_PATCH, p = idx % N_PATCH;
  float px1 = (float)(p % 7) * 32.f, py1 = (float)(p / 7) * 32.f;
  float ix = fminf(bb[n * 4 + 2], px1 + 32.f) - fmaxf(bb[n * 4 + 0], px1);
  float iy = fminf(bb[n * 4 + 3], py1 + 32.f) - fmaxf(bb[n * 4 + 1], py1);
  Mov[idx] = fmaxf(ix, 0.f) * fmaxf(iy, 0.f) * (1.f / 1024.f);
}

__global__ void init_x_k(const float* __restrict__ cls, const float* __restrict__ feat,
                         float* __restrict__ x) {
  int idx = blockIdx.x * 256 + threadIdx.x;
  if (idx >= TOKENS * DIM) return;
  int s = idx / (N_ROI * DIM);
  int d = idx % DIM;
  x[idx] = (s == 0) ? cls[d] : feat[(s - 1) * DIM + d];
}

extern "C" void kernel_launch(void* const* d_in, const int* in_sizes, int n_in,
                              void* d_out, int out_size, void* d_ws, size_t ws_size,
                              hipStream_t stream)
{
  (void)in_sizes; (void)n_in; (void)ws_size;
  const float* frame  = (const float*)d_in[0];
  const float* bboxes = (const float*)d_in[1];
  const float* conv_w = (const float*)d_in[2];
  const float* cls    = (const float*)d_in[3];
  const float* ln1_g  = (const float*)d_in[4];
  const float* ln1_b  = (const float*)d_in[5];
  const float* q_w = (const float*)d_in[6];  const float* q_b = (const float*)d_in[7];
  const float* k_w = (const float*)d_in[8];  const float* k_b = (const float*)d_in[9];
  const float* v_w = (const float*)d_in[10]; const float* v_b = (const float*)d_in[11];
  const float* o_w = (const float*)d_in[12]; const float* o_b = (const float*)d_in[13];
  const float* ln2_g = (const float*)d_in[14]; const float* ln2_b = (const float*)d_in[15];
  const float* fc_w  = (const float*)d_in[16]; const float* fc_b  = (const float*)d_in[17];
  const float* p_w   = (const float*)d_in[18]; const float* p_b   = (const float*)d_in[19];

  float* ws = (float*)d_ws;
  const long SX = (long)TOKENS * DIM;          // 2,457,600 floats
  float* x    = ws;                            // (SEQ, N_ROI, DIM)
  float* hbuf = ws + 1 * SX;
  float* qb_  = ws + 2 * SX;
  float* kb_  = ws + 3 * SX;
  float* vb_  = ws + 4 * SX;
  float* ob_  = ws + 5 * SX;
  float* ubuf = qb_;                           // aliases q..o : 4*SX == TOKENS*MLP_DIM
  float* feat = ws + 6 * SX;                   // 49*768
  float* Mov  = feat + N_PATCH * DIM;          // 64*49
  float* col  = Mov + N_ROI * N_PATCH;         // 49*3072

  im2col_k<<<(N_PATCH * 3072 + 255) / 256, 256, 0, stream>>>(frame, col);
  overlap_k<<<(N_ROI * N_PATCH + 255) / 256, 256, 0, stream>>>(bboxes, Mov);
  // patch embedding: feat = relu(col @ conv_w^T)   (M=49 padded inside one 128-row block)
  gemm_wmma<<<dim3(DIM / BN, 1), 256, 0, stream>>>(col, conv_w, nullptr, feat,
                                                   N_PATCH, DIM, 3072, 1, 0);
  init_x_k<<<(TOKENS * DIM + 255) / 256, 256, 0, stream>>>(cls, feat, x);

  dim3 g768(DIM / BN, TOKENS / BM);        // (12, 25)
  dim3 g3072(MLP_DIM / BN, TOKENS / BM);   // (48, 25)
  for (int l = 0; l < L_LAYERS; ++l) {
    layernorm_k<<<TOKENS, 256, 0, stream>>>(x, hbuf, ln1_g + l * DIM, ln1_b + l * DIM);
    gemm_wmma<<<g768, 256, 0, stream>>>(hbuf, q_w + (long)l * DIM * DIM, q_b + l * DIM,
                                        qb_, TOKENS, DIM, DIM, 0, 0);
    gemm_wmma<<<g768, 256, 0, stream>>>(hbuf, k_w + (long)l * DIM * DIM, k_b + l * DIM,
                                        kb_, TOKENS, DIM, DIM, 0, 0);
    gemm_wmma<<<g768, 256, 0, stream>>>(hbuf, v_w + (long)l * DIM * DIM, v_b + l * DIM,
                                        vb_, TOKENS, DIM, DIM, 0, 0);
    attn_diag_k<<<dim3(N_ROI, HEADS), 128, 0, stream>>>(qb_, kb_, vb_, Mov, ob_);
    gemm_wmma<<<g768, 256, 0, stream>>>(ob_, o_w + (long)l * DIM * DIM, o_b + l * DIM,
                                        x, TOKENS, DIM, DIM, 0, 1);
    layernorm_k<<<TOKENS, 256, 0, stream>>>(x, hbuf, ln2_g + l * DIM, ln2_b + l * DIM);
    gemm_wmma<<<g3072, 256, 0, stream>>>(hbuf, fc_w + (long)l * MLP_DIM * DIM,
                                         fc_b + l * MLP_DIM, ubuf,
                                         TOKENS, MLP_DIM, DIM, 2, 0);
    gemm_wmma<<<g768, 256, 0, stream>>>(ubuf, p_w + (long)l * DIM * MLP_DIM, p_b + l * DIM,
                                        x, TOKENS, DIM, MLP_DIM, 0, 1);
  }
  hipMemcpyAsync(d_out, x, (size_t)out_size * sizeof(float),
                 hipMemcpyDeviceToDevice, stream);
}
